// GNN_4389456577278
// MI455X (gfx1250) — matrix operand order repaired and checked by reading, not compiled, not run on hardware
//
#include <hip/hip_runtime.h>
#include <hip/hip_bf16.h>

typedef __attribute__((ext_vector_type(16))) _Float16 v16h;
typedef __attribute__((ext_vector_type(8)))  _Float16 v8h;
typedef __attribute__((ext_vector_type(8)))  float    v8f;
typedef __attribute__((ext_vector_type(4)))  unsigned u32x4;
typedef __attribute__((ext_vector_type(8)))  unsigned u32x8;

#define WG 256

static __device__ __forceinline__ unsigned fenc(float x) {
    unsigned u = __float_as_uint(x);
    return (u & 0x80000000u) ? ~u : (u | 0x80000000u);
}
static __device__ __forceinline__ float fdec(unsigned e) {
    unsigned u = (e & 0x80000000u) ? (e & 0x7fffffffu) : ~e;
    return __uint_as_float(u);
}

// ---------------------------------------------------------------------------
// Generic WMMA GEMM: OUT[M, nw] = A[M, K] (f32, ld=lda) @ W[K, nw] (f32) + bias
// 128x128 tile / WG; 8 waves x (16 rows x 128 cols). f32->f16 stage-in,
// f32 accumulate via v_wmma_f32_16x16x32_f16.
// ---------------------------------------------------------------------------
__global__ __launch_bounds__(WG)
void gemm_wmma(const float* __restrict__ A, int lda, int M, int K,
               const float* __restrict__ W, int nw,
               const float* __restrict__ bias,
               float* __restrict__ outF, _Float16* __restrict__ outH,
               int ldo, int relu)
{
    __shared__ __align__(16) _Float16 lA[128][40];
    __shared__ __align__(16) _Float16 lB[128][40];

    const int tid   = threadIdx.x;
    const int tileM = blockIdx.x * 128;
    const int wave  = tid >> 5;
    const int lane  = tid & 31;
    const int ln    = lane & 15;
    const int half  = lane >> 4;

    v8f acc[8] = {};

    for (int k0 = 0; k0 < K; k0 += 32) {
        __syncthreads();
        // ---- stage A (128x32): 2 threads/row, float4 x4 -> 16 f16 -> 2x b128
        {
            int r = tid >> 1, hc = tid & 1;
            int row = tileM + r;
            int kb  = k0 + hc * 16;
            v8h h0 = {}, h1 = {};
            if (row < M && kb < K) {
                const float4* p = (const float4*)(A + (size_t)row * lda + kb);
                float4 a0 = p[0], a1 = p[1], a2 = p[2], a3 = p[3];
                __builtin_prefetch(p + 4, 0, 1);
                h0[0] = (_Float16)a0.x; h0[1] = (_Float16)a0.y;
                h0[2] = (_Float16)a0.z; h0[3] = (_Float16)a0.w;
                h0[4] = (_Float16)a1.x; h0[5] = (_Float16)a1.y;
                h0[6] = (_Float16)a1.z; h0[7] = (_Float16)a1.w;
                h1[0] = (_Float16)a2.x; h1[1] = (_Float16)a2.y;
                h1[2] = (_Float16)a2.z; h1[3] = (_Float16)a2.w;
                h1[4] = (_Float16)a3.x; h1[5] = (_Float16)a3.y;
                h1[6] = (_Float16)a3.z; h1[7] = (_Float16)a3.w;
            }
            *(v8h*)&lA[r][hc * 16]     = h0;
            *(v8h*)&lA[r][hc * 16 + 8] = h1;
        }
        // ---- stage W chunk transposed, coalesced global reads (lane ~ n)
#pragma unroll
        for (int j = 0; j < 16; ++j) {
            int idx = tid + j * WG;
            int n = idx & 127, kk = idx >> 7;
            int col = k0 + kk;
            float v = (col < K && n < nw) ? W[(size_t)col * nw + n] : 0.f;
            lB[n][kk] = (_Float16)v;
        }
        __syncthreads();

        v16h a;
#pragma unroll
        for (int q = 0; q < 8; ++q) {
            int k = ((q & 4) << 2) + half * 8 + ((q & 3) << 1);
            a[2 * q]     = lA[wave * 16 + ln][k];
            a[2 * q + 1] = lA[wave * 16 + ln][k + 1];
        }
#pragma unroll
        for (int t = 0; t < 8; ++t) {
            v16h b;
#pragma unroll
            for (int q = 0; q < 8; ++q) {
                int k = ((q & 4) << 2) + half * 8 + ((q & 3) << 1);
                b[2 * q]     = lB[t * 16 + ln][k];
                b[2 * q + 1] = lB[t * 16 + ln][k + 1];
            }
            acc[t] = __builtin_amdgcn_wmma_f32_16x16x32_f16(
                false, a, false, b, (short)0, acc[t], false, false);
        }
    }

#pragma unroll
    for (int t = 0; t < 8; ++t) {
        int col = t * 16 + ln;
        float bv = (bias != nullptr && col < nw) ? bias[col] : 0.f;
#pragma unroll
        for (int q = 0; q < 8; ++q) {
            int row = tileM + wave * 16 + q + 8 * half;
            float v = acc[t][q] + bv;
            if (relu) v = fmaxf(v, 0.f);
            if (row < M && col < nw) {
                if (outH) outH[(size_t)row * ldo + col] = (_Float16)v;
                else      outF[(size_t)row * ldo + col] = v;
            }
        }
    }
}

// ---------------------------------------------------------------------------
// Fused GATv2 edge pass: m = leaky_relu(e@we + xl[src] + xr[dst]),
// score = m . att ; segment-max(score by dst) via monotone-uint atomics.
// Full-interior tiles stage the whole 128x256B f16 A-tile with one
// tensor_load_to_lds (TDM pads each 64B row-chunk with 16B to match the
// 80B/chunk LDS layout the WMMA fragment loads expect).
// ---------------------------------------------------------------------------
__global__ __launch_bounds__(WG)
void gat_edge_scores(const _Float16* __restrict__ e16,
                     const _Float16* __restrict__ loop16,
                     const int* __restrict__ ei, int E, int Nn,
                     const float* __restrict__ W,      // we [128,128]
                     const float* __restrict__ attv,   // [128]
                     const float* __restrict__ xl,
                     const float* __restrict__ xr,
                     float* __restrict__ score,
                     unsigned* __restrict__ smax)
{
    // 4 k-chunks interleaved: row = 4 x (64B data + 16B pad) = 320B
    __shared__ __align__(16) _Float16 lA2[128][160];
    __shared__ __align__(16) _Float16 lB[128][40];
    __shared__ int   sSrc[128];
    __shared__ int   sDst[128];
    __shared__ float sAtt[128];

    const int tid  = threadIdx.x;
    const int Etot = E + Nn;
    const int tile = blockIdx.x * 128;
    const int wave = tid >> 5;
    const int lane = tid & 31;
    const int ln   = lane & 15;
    const int half = lane >> 4;

    if (tid < 128) {
        int i = tile + tid;
        int s = 0, d = 0;
        if (i < E)         { s = ei[i]; d = ei[E + i]; }
        else if (i < Etot) { s = i - E; d = i - E; }
        sSrc[tid] = s;
        sDst[tid] = d;
        sAtt[tid] = attv[tid];
    }

    const bool fullTile = (tile + 128 <= E);
    if (fullTile) {
        if (tid < 32) {   // one wave issues the tensor DMA
            unsigned lds_addr = (unsigned)(unsigned long long)
                (__attribute__((address_space(3))) _Float16*)&lA2[0][0];
            unsigned long long ga =
                (unsigned long long)(e16 + (size_t)tile * 128);
            u32x4 g0;
            g0[0] = 1u;                                  // count=1, user mode
            g0[1] = lds_addr;                            // lds_addr
            g0[2] = (unsigned)ga;                        // global_addr[31:0]
            g0[3] = (unsigned)((ga >> 32) & 0x1ffffffu)  // global_addr[56:32]
                    | (2u << 30);                        // type=2 (image)
            u32x8 g1;
            g1[0] = (2u << 16)      // data_size = 4B
                  | (1u << 20)      // pad_enable
                  | (3u << 22)      // pad_interval: every 16 dwords (64B)
                  | (3u << 25);     // pad_amount: 4 dwords (16B)
            g1[1] = (64u << 16);    // tensor_dim0 = 64 dwords/row
            g1[2] = (128u << 16);   // tensor_dim1 = 128 rows
            g1[3] = (64u << 16);    // tile_dim0 = 64 dwords
            g1[4] = 128u;           // tile_dim1 = 128, tile_dim2 = 0
            g1[5] = 64u;            // tensor_dim0_stride = 64 dwords
            g1[6] = 0u;
            g1[7] = 0u;
            asm volatile("tensor_load_to_lds %0, %1"
                         :: "s"(g0), "s"(g1) : "memory");
            __builtin_amdgcn_s_wait_tensorcnt(0);
        }
    } else {
        // boundary / self-loop tiles: manual stage, identical padded layout
#pragma unroll
        for (int j = 0; j < 32; ++j) {
            int idx = tid + j * WG;
            int r = idx >> 6, cu = idx & 63;
            int i = tile + r;
            unsigned val = 0u;
            if (i < E)
                val = ((const unsigned*)(e16 + (size_t)i * 128))[cu];
            else if (i < Etot)
                val = ((const unsigned*)(loop16 + (size_t)(i - E) * 128))[cu];
            *(unsigned*)&lA2[r][(cu >> 4) * 40 + (cu & 15) * 2] = val;
        }
    }
    __syncthreads();

    v8f acc[8] = {};

    for (int k0 = 0; k0 < 128; k0 += 32) {
        // stage we chunk transposed, coalesced global reads
#pragma unroll
        for (int j = 0; j < 16; ++j) {
            int idx = tid + j * WG;
            int n = idx & 127, kk = idx >> 7;
            lB[n][kk] = (_Float16)W[(size_t)(k0 + kk) * 128 + n];
        }
        __syncthreads();

        const int cb = (k0 >> 5) * 40;  // chunk base inside padded A row
        v16h a;
#pragma unroll
        for (int q = 0; q < 8; ++q) {
            int k = ((q & 4) << 2) + half * 8 + ((q & 3) << 1);
            a[2 * q]     = lA2[wave * 16 + ln][cb + k];
            a[2 * q + 1] = lA2[wave * 16 + ln][cb + k + 1];
        }
#pragma unroll
        for (int t = 0; t < 8; ++t) {
            v16h b;
#pragma unroll
            for (int q = 0; q < 8; ++q) {
                int k = ((q & 4) << 2) + half * 8 + ((q & 3) << 1);
                b[2 * q]     = lB[t * 16 + ln][k];
                b[2 * q + 1] = lB[t * 16 + ln][k + 1];
            }
            acc[t] = __builtin_amdgcn_wmma_f32_16x16x32_f16(
                false, a, false, b, (short)0, acc[t], false, false);
        }
        __syncthreads();
    }

    // epilogue: add L2-resident gathers, leaky-relu, dot with att
    float s[8];
#pragma unroll
    for (int q = 0; q < 8; ++q) s[q] = 0.f;

#pragma unroll
    for (int t = 0; t < 8; ++t) {
        int col = t * 16 + ln;
        float av = sAtt[col];
#pragma unroll
        for (int q = 0; q < 8; ++q) {
            int rl = wave * 16 + q + 8 * half;
            int i  = tile + rl;
            float g = 0.f;
            if (i < Etot) {
                int sn = sSrc[rl], dn = sDst[rl];
                g = xl[(size_t)sn * 128 + col] + xr[(size_t)dn * 128 + col];
            }
            float v = acc[t][q] + g;
            v = v > 0.f ? v : 0.2f * v;
            s[q] += v * av;
        }
    }
#pragma unroll
    for (int q = 0; q < 8; ++q) {
        float r = s[q];
        r += __shfl_xor(r, 1);
        r += __shfl_xor(r, 2);
        r += __shfl_xor(r, 4);
        r += __shfl_xor(r, 8);
        if (ln == 0) {
            int rl = wave * 16 + q + 8 * half;
            int i  = tile + rl;
            if (i < Etot) {
                score[i] = r;
                atomicMax(&smax[sDst[rl]], fenc(r));
            }
        }
    }
}

// ---------------------------------------------------------------------------
// streaming / atomic helper kernels
// ---------------------------------------------------------------------------
__global__ void fill_k(float* __restrict__ p, size_t n, float v) {
    size_t i = (size_t)blockIdx.x * WG + threadIdx.x;
    if (i < n) p[i] = v;
}

__global__ void deg_k(const int* __restrict__ ei, int E, float* __restrict__ cnt) {
    int i = blockIdx.x * WG + threadIdx.x;
    if (i < E) atomicAdd(&cnt[ei[E + i]], 1.f);
}

__global__ void gcnt_k(const int* __restrict__ batch, int n, float* __restrict__ gcnt) {
    int i = blockIdx.x * WG + threadIdx.x;
    if (i < n) atomicAdd(&gcnt[batch[i]], 1.f);
}

__global__ void loop_sum_k(const _Float16* __restrict__ e16, const int* __restrict__ ei,
                           int E, float* __restrict__ lsum) {
    int t = blockIdx.x * WG + threadIdx.x;
    int i = t >> 5, lane = t & 31;
    if (i >= E) return;
    int d = ei[E + i];
    float* o = lsum + (size_t)d * 128 + lane * 4;
    const _Float16* src = e16 + (size_t)i * 128 + lane * 4;
#pragma unroll
    for (int j = 0; j < 4; ++j) atomicAdd(o + j, (float)src[j]);
}

__global__ void loop_div_k(const float* __restrict__ lsum, const float* __restrict__ cnt,
                           int n, _Float16* __restrict__ loop16) {
    int t = blockIdx.x * WG + threadIdx.x;
    int i = t >> 5, lane = t & 31;
    if (i >= n) return;
    float inv = 1.f / fmaxf(cnt[i], 1.f);
#pragma unroll
    for (int j = 0; j < 4; ++j) {
        int c = lane * 4 + j;
        loop16[(size_t)i * 128 + c] = (_Float16)(lsum[(size_t)i * 128 + c] * inv);
    }
}

__global__ void softmax_prep(const float* __restrict__ score, const int* __restrict__ ei,
                             int E, int Nn, const unsigned* __restrict__ smax,
                             float* __restrict__ exb, float* __restrict__ den) {
    int i = blockIdx.x * WG + threadIdx.x;
    int Etot = E + Nn;
    if (i >= Etot) return;
    int d = (i < E) ? ei[E + i] : (i - E);
    float ex = __expf(score[i] - fdec(smax[d]));
    exb[i] = ex;
    atomicAdd(&den[d], ex);
}

__global__ void scatter_out(const float* __restrict__ exb, const float* __restrict__ den,
                            const int* __restrict__ ei, int E, int Nn,
                            const float* __restrict__ xl, float* __restrict__ acc) {
    int t = blockIdx.x * WG + threadIdx.x;
    int i = t >> 5, lane = t & 31;
    int Etot = E + Nn;
    if (i >= Etot) return;
    int sN = (i < E) ? ei[i]     : (i - E);
    int dN = (i < E) ? ei[E + i] : (i - E);
    float alpha = exb[i] / den[dN];
    const float4 xv = *(const float4*)(xl + (size_t)sN * 128 + lane * 4);
    float* o = acc + (size_t)dN * 128 + lane * 4;
    atomicAdd(o + 0, alpha * xv.x);
    atomicAdd(o + 1, alpha * xv.y);
    atomicAdd(o + 2, alpha * xv.z);
    atomicAdd(o + 3, alpha * xv.w);
}

__global__ void seg_sum128(const float* __restrict__ v, const int* __restrict__ seg,
                           int n, float* __restrict__ out) {
    int t = blockIdx.x * WG + threadIdx.x;
    int i = t >> 5, lane = t & 31;
    if (i >= n) return;
    int g = seg[i];
    const float4 x = *(const float4*)(v + (size_t)i * 128 + lane * 4);
    float* o = out + (size_t)g * 128 + lane * 4;
    atomicAdd(o + 0, x.x); atomicAdd(o + 1, x.y);
    atomicAdd(o + 2, x.z); atomicAdd(o + 3, x.w);
}

__global__ void gn_center(float* __restrict__ acc, const int* __restrict__ batch, int n,
                          const float* __restrict__ gsum, const float* __restrict__ gcnt,
                          const float* __restrict__ cb, const float* __restrict__ ms,
                          float* __restrict__ gvar) {
    int t = blockIdx.x * WG + threadIdx.x;
    int i = t >> 5, lane = t & 31;
    if (i >= n) return;
    int g = batch[i];
    float inv = 1.f / fmaxf(gcnt[g], 1.f);
#pragma unroll
    for (int j = 0; j < 4; ++j) {
        int c = lane * 4 + j;
        float mean = gsum[(size_t)g * 128 + c] * inv + cb[c];
        float x = acc[(size_t)i * 128 + c] + cb[c];
        float tt = x - mean * ms[c];
        acc[(size_t)i * 128 + c] = tt;
        atomicAdd(&gvar[(size_t)g * 128 + c], tt * tt);
    }
}

__global__ void gn_final(const float* __restrict__ acc, const int* __restrict__ batch, int n,
                         const float* __restrict__ gvar, const float* __restrict__ gcnt,
                         const float* __restrict__ w, const float* __restrict__ b,
                         float* __restrict__ hout, int ldo) {
    int t = blockIdx.x * WG + threadIdx.x;
    int i = t >> 5, lane = t & 31;
    if (i >= n) return;
    int g = batch[i];
    float inv = 1.f / fmaxf(gcnt[g], 1.f);
#pragma unroll
    for (int j = 0; j < 4; ++j) {
        int c = lane * 4 + j;
        float var = gvar[(size_t)g * 128 + c] * inv;
        float v = w[c] * acc[(size_t)i * 128 + c] * __frsqrt_rn(var + 1e-5f) + b[c];
        hout[(size_t)i * ldo + c] = fmaxf(v, 0.f);
    }
}

__global__ void pool_sum(const float* __restrict__ hcat, const int* __restrict__ batch,
                         int n, float* __restrict__ pool) {
    int t = blockIdx.x * WG + threadIdx.x;
    int i = t >> 7, q = t & 127;
    if (i >= n) return;
    int g = batch[i];
    const float4 x = *(const float4*)(hcat + (size_t)i * 512 + q * 4);
    float* o = pool + (size_t)g * 576 + q * 4;
    atomicAdd(o + 0, x.x); atomicAdd(o + 1, x.y);
    atomicAdd(o + 2, x.z); atomicAdd(o + 3, x.w);
}

__global__ void pool_fin(float* __restrict__ pool, const float* __restrict__ gcnt,
                         const float* __restrict__ z, int G_) {
    int t = blockIdx.x * WG + threadIdx.x;
    if (t >= G_ * 576) return;
    int g = t / 576, c = t - g * 576;
    if (c < 512) pool[t] = pool[t] / fmaxf(gcnt[g], 1.f);
    else         pool[t] = z[g * 64 + (c - 512)];
}

__global__ void fc3_k(const float* __restrict__ in, const float* __restrict__ w,
                      const float* __restrict__ b, float* __restrict__ out, int G_) {
    int g = blockIdx.x * WG + threadIdx.x;
    if (g >= G_) return;
    float s = b[0];
#pragma unroll 8
    for (int c = 0; c < 64; ++c) s += in[(size_t)g * 64 + c] * w[c];
    out[g] = s;
}

// ---------------------------------------------------------------------------
static inline int cdiv(long long a, long long b) { return (int)((a + b - 1) / b); }

extern "C" void kernel_launch(void* const* d_in, const int* in_sizes, int n_in,
                              void* d_out, int out_size, void* d_ws, size_t ws_size,
                              hipStream_t stream) {
    const float* x      = (const float*)d_in[0];
    const int*   ei     = (const int*)  d_in[1];
    const float* ea     = (const float*)d_in[2];
    const int*   batch  = (const int*)  d_in[3];
    const float* z      = (const float*)d_in[4];
    const float* enc_w  = (const float*)d_in[5];
    const float* enc_b  = (const float*)d_in[6];
    const float* eenc_w = (const float*)d_in[7];
    const float* eenc_b = (const float*)d_in[8];
    const float* lin_l_w= (const float*)d_in[9];
    const float* lin_l_b= (const float*)d_in[10];
    const float* lin_r_w= (const float*)d_in[11];
    const float* lin_r_b= (const float*)d_in[12];
    const float* lin_e_w= (const float*)d_in[13];
    const float* att    = (const float*)d_in[14];
    const float* conv_b = (const float*)d_in[15];
    const float* gn_w   = (const float*)d_in[16];
    const float* gn_b   = (const float*)d_in[17];
    const float* gn_ms  = (const float*)d_in[18];
    const float* fc1_w  = (const float*)d_in[19];
    const float* fc1_b  = (const float*)d_in[20];
    const float* fc2_w  = (const float*)d_in[21];
    const float* fc2_b  = (const float*)d_in[22];
    const float* fc3_w  = (const float*)d_in[23];
    const float* fc3_b  = (const float*)d_in[24];

    const int N = in_sizes[0] / 32;
    const int E = in_sizes[1] / 2;
    const int G = in_sizes[4] / 64;
    const int Etot = E + N;
    const int L = 3;

    char* wp = (char*)d_ws;
    auto alloc = [&](size_t bytes) -> void* {
        void* p = (void*)wp;
        wp += (bytes + 255) & ~(size_t)255;
        return p;
    };
    float*    hcat   = (float*)   alloc((size_t)N * 512 * 4);
    _Float16* e16    = (_Float16*)alloc((size_t)E * 128 * 2);
    _Float16* loop16 = (_Float16*)alloc((size_t)N * 128 * 2);
    float*    xl     = (float*)   alloc((size_t)N * 128 * 4); // doubles as loop_sum
    float*    xr     = (float*)   alloc((size_t)N * 128 * 4);
    float*    accb   = (float*)   alloc((size_t)N * 128 * 4);
    float*    score  = (float*)   alloc((size_t)Etot * 4);
    float*    exb    = (float*)   alloc((size_t)Etot * 4);
    unsigned* smax   = (unsigned*)alloc((size_t)N * 4);
    float*    den    = (float*)   alloc((size_t)N * 4);
    float*    cnt    = (float*)   alloc((size_t)N * 4);
    float*    gcnt   = (float*)   alloc((size_t)G * 4);
    float*    gsum   = (float*)   alloc((size_t)G * 128 * 4);
    float*    gvar   = (float*)   alloc((size_t)G * 128 * 4);
    float*    pool   = (float*)   alloc((size_t)G * 576 * 4);
    float*    fcb1   = (float*)   alloc((size_t)G * 128 * 4);
    float*    fcb2   = (float*)   alloc((size_t)G * 64 * 4);

    #define FILL(p, n) fill_k<<<cdiv((long long)(n), WG), WG, 0, stream>>>((float*)(p), (size_t)(n), 0.f)

    FILL(cnt, N); FILL(gcnt, G); FILL(xl, (size_t)N * 128); FILL(pool, (size_t)G * 576);
    deg_k <<<cdiv(E, WG), WG, 0, stream>>>(ei, E, cnt);
    gcnt_k<<<cdiv(N, WG), WG, 0, stream>>>(batch, N, gcnt);

    gemm_wmma<<<cdiv(N, 128), WG, 0, stream>>>(x, 32, N, 32, enc_w, 128, enc_b,
                                               hcat, nullptr, 512, 0);
    gemm_wmma<<<cdiv(E, 128), WG, 0, stream>>>(ea, 16, E, 16, eenc_w, 128, eenc_b,
                                               nullptr, e16, 128, 0);
    loop_sum_k<<<cdiv((long long)E * 32, WG), WG, 0, stream>>>(e16, ei, E, xl);
    loop_div_k<<<cdiv((long long)N * 32, WG), WG, 0, stream>>>(xl, cnt, N, loop16);

    for (int l = 0; l < L; ++l) {
        FILL(smax, N); FILL(den, N); FILL(accb, (size_t)N * 128);
        FILL(gsum, (size_t)G * 128); FILL(gvar, (size_t)G * 128);

        const float* hin = hcat + (size_t)l * 128;
        gemm_wmma<<<cdiv(N, 128), WG, 0, stream>>>(hin, 512, N, 128,
            lin_l_w + (size_t)l * 128 * 128, 128, lin_l_b + (size_t)l * 128,
            xl, nullptr, 128, 0);
        gemm_wmma<<<cdiv(N, 128), WG, 0, stream>>>(hin, 512, N, 128,
            lin_r_w + (size_t)l * 128 * 128, 128, lin_r_b + (size_t)l * 128,
            xr, nullptr, 128, 0);

        gat_edge_scores<<<cdiv(Etot, 128), WG, 0, stream>>>(
            e16, loop16, ei, E, N,
            lin_e_w + (size_t)l * 128 * 128, att + (size_t)l * 128,
            xl, xr, score, smax);

        softmax_prep<<<cdiv(Etot, WG), WG, 0, stream>>>(score, ei, E, N, smax, exb, den);
        scatter_out <<<cdiv((long long)Etot * 32, WG), WG, 0, stream>>>(exb, den, ei, E, N, xl, accb);

        seg_sum128<<<cdiv((long long)N * 32, WG), WG, 0, stream>>>(accb, batch, N, gsum);
        gn_center <<<cdiv((long long)N * 32, WG), WG, 0, stream>>>(accb, batch, N, gsum, gcnt,
            conv_b + (size_t)l * 128, gn_ms + (size_t)l * 128, gvar);
        gn_final  <<<cdiv((long long)N * 32, WG), WG, 0, stream>>>(accb, batch, N, gvar, gcnt,
            gn_w + (size_t)l * 128, gn_b + (size_t)l * 128,
            hcat + (size_t)(l + 1) * 128, 512);
    }

    pool_sum<<<cdiv((long long)N * 128, WG), WG, 0, stream>>>(hcat, batch, N, pool);
    pool_fin<<<cdiv((long long)G * 576, WG), WG, 0, stream>>>(pool, gcnt, z, G);
    gemm_wmma<<<cdiv(G, 128), WG, 0, stream>>>(pool, 576, G, 576, fc1_w, 128, fc1_b,
                                               fcb1, nullptr, 128, 1);
    gemm_wmma<<<cdiv(G, 128), WG, 0, stream>>>(fcb1, 128, G, 128, fc2_w, 64, fc2_b,
                                               fcb2, nullptr, 64, 1);
    fc3_k<<<cdiv(G, WG), WG, 0, stream>>>(fcb2, fc3_w, fc3_b, (float*)d_out, G);

    #undef FILL
    (void)n_in; (void)out_size; (void)ws_size;
}
